// PatchCore_62826781605950
// MI455X (gfx1250) — matrix operand order repaired and checked by reading, not compiled, not run on hardware
//
#include <hip/hip_runtime.h>
#include <math.h>

// ---------------------------------------------------------------------------
// PatchCore on MI455X (gfx1250, wave32):
//   dists = sqrt(clip(|a|^2 - 2ab + |b|^2, 0));  patch_scores = min over M
//   image_scores = max per image;  masks = blur(resize(patch_scores))
// Main cost: 50000x3136x1024 f32 GEMM.
// Strategy: f32 -> bf16 hi/lo split, ab ~= a_hi*b_hi + a_hi*b_lo + a_lo*b_hi
//           via v_wmma_f32_16x16x32_bf16. Splits are PRECOMPUTED into the
//           workspace (B always ~13MB; A ~205MB if ws_size allows) so the
//           WMMA loop is load+WMMA only, not VALU-conversion-bound.
// ---------------------------------------------------------------------------

typedef __attribute__((ext_vector_type(16))) __bf16 v16bf;
typedef __attribute__((ext_vector_type(8)))  __bf16 v8bf;
typedef __attribute__((ext_vector_type(4)))  __bf16 v4bf;
typedef __attribute__((ext_vector_type(8)))  float  v8f;

#define D_DIM 1024
#define OH 224
#define OW 224
#define GH 28
#define GW 28
#define NB 4
#define RADIUS 16

// Convert 4 f32 -> bf16 hi + residual lo at compile-time base offset B.
#define CVT4(F, HI, LO, B) do {                                          \
    __bf16 h0 = (__bf16)(F).x; (HI)[(B)+0] = h0;                         \
    (LO)[(B)+0] = (__bf16)((F).x - (float)h0);                           \
    __bf16 h1 = (__bf16)(F).y; (HI)[(B)+1] = h1;                         \
    (LO)[(B)+1] = (__bf16)((F).y - (float)h1);                           \
    __bf16 h2 = (__bf16)(F).z; (HI)[(B)+2] = h2;                         \
    (LO)[(B)+2] = (__bf16)((F).z - (float)h2);                           \
    __bf16 h3 = (__bf16)(F).w; (HI)[(B)+3] = h3;                         \
    (LO)[(B)+3] = (__bf16)((F).w - (float)h3);                           \
} while (0)

#define SHUF16(A, B) __builtin_shufflevector((A), (B), \
    0, 1, 2, 3, 4, 5, 6, 7, 8, 9, 10, 11, 12, 13, 14, 15)

// ---------------- f32 -> (hi, lo) bf16 split, 4 elements/thread ----------------
__global__ void split_bf16_kernel(const float* __restrict__ x,
                                  __bf16* __restrict__ hi,
                                  __bf16* __restrict__ lo, int n4) {
    int i = blockIdx.x * blockDim.x + threadIdx.x;
    if (i >= n4) return;
    float4 f = ((const float4*)x)[i];
    __bf16 h0 = (__bf16)f.x, h1 = (__bf16)f.y, h2 = (__bf16)f.z, h3 = (__bf16)f.w;
    v4bf hv = {h0, h1, h2, h3};
    v4bf lv = {(__bf16)(f.x - (float)h0), (__bf16)(f.y - (float)h1),
               (__bf16)(f.z - (float)h2), (__bf16)(f.w - (float)h3)};
    ((v4bf*)hi)[i] = hv;
    ((v4bf*)lo)[i] = lv;
}

// ---------------- row squared-norm: one wave per row ----------------
__global__ void rownorm_kernel(const float* __restrict__ x,
                               float* __restrict__ out, int rows) {
    int gtid = blockIdx.x * blockDim.x + threadIdx.x;
    int wave = gtid >> 5;
    int lane = gtid & 31;
    if (wave >= rows) return;
    const float* r = x + (size_t)wave * D_DIM;
    float s = 0.0f;
    for (int i = lane; i < D_DIM; i += 32) { float v = r[i]; s += v * v; }
    for (int off = 16; off > 0; off >>= 1) s += __shfl_xor(s, off, 32);
    if (lane == 0) out[wave] = s;
}

// ---------------- init per-q min to +inf bits ----------------
__global__ void init_min_kernel(unsigned* __restrict__ ps, int Q) {
    int i = blockIdx.x * blockDim.x + threadIdx.x;
    if (i < Q) ps[i] = 0x7F800000u;   // +inf
}

// ---------------- gaussian weights (sigma=4, radius=16) ----------------
__global__ void gauss_kernel(float* __restrict__ w) {
    if (blockIdx.x == 0 && threadIdx.x == 0) {
        float tmp[2 * RADIUS + 1];
        float s = 0.0f;
        for (int i = 0; i <= 2 * RADIUS; ++i) {
            float x = (float)(i - RADIUS) / 4.0f;
            tmp[i] = expf(-0.5f * x * x);
            s += tmp[i];
        }
        for (int i = 0; i <= 2 * RADIUS; ++i) w[i] = tmp[i] / s;
    }
}

// ---------------- main GEMM + min reduction ----------------
// Workgroup: 256 threads (8 waves). WG tile = 256(M) x 64(Q).
// Wave tile: 32(M) x 64(Q) = 2 x 4 WMMA C-tiles, K stepped by 32.
// MODE 2: A and B pre-split to bf16 hi/lo (pure load + 24 WMMA per K-step)
// MODE 1: B pre-split, A converted in-registers
// MODE 0: both converted in-registers (tiny-workspace fallback)
template <int MODE>
__global__ __launch_bounds__(256)
void gemm_min_kernel(const float*  __restrict__ Af,   // coreset  [M,1024] f32
                     const float*  __restrict__ Bf,   // features [Q,1024] f32
                     const __bf16* __restrict__ Ahi, const __bf16* __restrict__ Alo,
                     const __bf16* __restrict__ Bhi, const __bf16* __restrict__ Blo,
                     const float*  __restrict__ a2,   // [M]
                     const float*  __restrict__ b2,   // [Q]
                     unsigned*     __restrict__ psmin, // [Q] f32-as-u32 min
                     int M) {
    __shared__ unsigned smin[64];
    const int tid = threadIdx.x;
    if (tid < 64) smin[tid] = 0x7F800000u;
    __syncthreads();

    const int lane  = tid & 31;
    const int wave  = tid >> 5;
    const int half  = lane >> 4;   // 0: lanes 0-15, 1: lanes 16-31
    const int l16   = lane & 15;
    const int qbase = blockIdx.x * 64;
    const int mbase = blockIdx.y * 256 + wave * 32;

    v8f zero = {};
    v8f acc[2][4];
#pragma unroll
    for (int ms = 0; ms < 2; ++ms)
#pragma unroll
        for (int ns = 0; ns < 4; ++ns) acc[ms][ns] = zero;

    // ---- per-lane row bases (A rows clamped for the M tail: dup is min-safe) ----
    int arowidx[2];
#pragma unroll
    for (int ms = 0; ms < 2; ++ms) {
        int m = mbase + ms * 16 + l16;
        if (m > M - 1) m = M - 1;
        arowidx[ms] = m;
    }
    const float*  arowf[2];
    const __bf16* arowh[2];
    const __bf16* arowl[2];
#pragma unroll
    for (int ms = 0; ms < 2; ++ms) {
        if constexpr (MODE == 2) {
            arowh[ms] = Ahi + (size_t)arowidx[ms] * D_DIM;
            arowl[ms] = Alo + (size_t)arowidx[ms] * D_DIM;
        } else {
            arowf[ms] = Af + (size_t)arowidx[ms] * D_DIM;
        }
    }
    const float*  browf[4];
    const __bf16* browh[4];
    const __bf16* browl[4];
#pragma unroll
    for (int ns = 0; ns < 4; ++ns) {
        int q = qbase + ns * 16 + l16;
        if constexpr (MODE >= 1) {
            browh[ns] = Bhi + (size_t)q * D_DIM;
            browl[ns] = Blo + (size_t)q * D_DIM;
        } else {
            browf[ns] = Bf + (size_t)q * D_DIM;
        }
    }

    for (int k0 = 0; k0 < D_DIM; k0 += 32) {
        // prefetch next K-step of the streamed coreset rows (global_prefetch_b8)
        if (k0 + 32 < D_DIM) {
            if constexpr (MODE == 2) {
                __builtin_prefetch(arowh[0] + k0 + 32, 0, 0);
                __builtin_prefetch(arowh[1] + k0 + 32, 0, 0);
                __builtin_prefetch(arowl[0] + k0 + 32, 0, 0);
                __builtin_prefetch(arowl[1] + k0 + 32, 0, 0);
            } else {
                __builtin_prefetch(arowf[0] + k0 + 32, 0, 0);
                __builtin_prefetch(arowf[1] + k0 + 32, 0, 0);
            }
        }

        // ---- A fragments (16x32 bf16: lane<16 K=0..15, lane>=16 K=16..31) ----
        v16bf ahi[2], alo[2];
#pragma unroll
        for (int ms = 0; ms < 2; ++ms) {
            if constexpr (MODE == 2) {
                const __bf16* ph = arowh[ms] + k0 + half * 16;
                const __bf16* pl = arowl[ms] + k0 + half * 16;
                v8bf h0 = *(const v8bf*)(ph);
                v8bf h1 = *(const v8bf*)(ph + 8);
                v8bf l0 = *(const v8bf*)(pl);
                v8bf l1 = *(const v8bf*)(pl + 8);
                ahi[ms] = SHUF16(h0, h1);
                alo[ms] = SHUF16(l0, l1);
            } else {
                const float* p = arowf[ms] + k0 + half * 16;
                float4 f0 = *(const float4*)(p + 0);
                float4 f1 = *(const float4*)(p + 4);
                float4 f2 = *(const float4*)(p + 8);
                float4 f3 = *(const float4*)(p + 12);
                CVT4(f0, ahi[ms], alo[ms], 0);
                CVT4(f1, ahi[ms], alo[ms], 4);
                CVT4(f2, ahi[ms], alo[ms], 8);
                CVT4(f3, ahi[ms], alo[ms], 12);
            }
        }
        // ---- B fragments (32x16 bf16: lane = column q, lane<16 K=0..15) ----
        v16bf bhi[4], blo[4];
#pragma unroll
        for (int ns = 0; ns < 4; ++ns) {
            if constexpr (MODE >= 1) {
                bhi[ns] = *(const v16bf*)(browh[ns] + k0 + half * 16);
                blo[ns] = *(const v16bf*)(browl[ns] + k0 + half * 16);
            } else {
                const float* p = browf[ns] + k0 + half * 16;
                float4 f0 = *(const float4*)(p + 0);
                float4 f1 = *(const float4*)(p + 4);
                float4 f2 = *(const float4*)(p + 8);
                float4 f3 = *(const float4*)(p + 12);
                CVT4(f0, bhi[ns], blo[ns], 0);
                CVT4(f1, bhi[ns], blo[ns], 4);
                CVT4(f2, bhi[ns], blo[ns], 8);
                CVT4(f3, bhi[ns], blo[ns], 12);
            }
        }
        // ---- 24 WMMAs per K-step ----
#pragma unroll
        for (int ms = 0; ms < 2; ++ms)
#pragma unroll
            for (int ns = 0; ns < 4; ++ns) {
                acc[ms][ns] = __builtin_amdgcn_wmma_f32_16x16x32_bf16(
                    false, ahi[ms], false, bhi[ns], (short)0, acc[ms][ns], false, false);
                acc[ms][ns] = __builtin_amdgcn_wmma_f32_16x16x32_bf16(
                    false, ahi[ms], false, blo[ns], (short)0, acc[ms][ns], false, false);
                acc[ms][ns] = __builtin_amdgcn_wmma_f32_16x16x32_bf16(
                    false, alo[ms], false, bhi[ns], (short)0, acc[ms][ns], false, false);
            }
    }

    // ---- epilogue: d2 = a2 - 2ab + b2, clamp, min over the 32 rows ----
    float a2v[2][8];
#pragma unroll
    for (int ms = 0; ms < 2; ++ms)
#pragma unroll
        for (int r = 0; r < 8; ++r) {
            int m = mbase + ms * 16 + half * 8 + r;   // C layout: VGPR r -> M = r + half*8
            if (m > M - 1) m = M - 1;
            a2v[ms][r] = a2[m];
        }
#pragma unroll
    for (int ns = 0; ns < 4; ++ns) {
        int q = qbase + ns * 16 + l16;
        float s = 3.4e38f;
#pragma unroll
        for (int ms = 0; ms < 2; ++ms)
#pragma unroll
            for (int r = 0; r < 8; ++r)
                s = fminf(s, a2v[ms][r] - 2.0f * acc[ms][ns][r]);
        s += b2[q];
        s = fmaxf(s, 0.0f);
        // lanes L and L^16 hold the same q with different M halves
        s = fminf(s, __shfl_xor(s, 16, 32));
        if (half == 0)
            atomicMin(&smin[ns * 16 + l16], __float_as_uint(s));  // ds_min_u32
    }
    __syncthreads();
    if (tid < 64)
        atomicMin(&psmin[qbase + tid], smin[tid]);  // global_atomic_min_u32
}

// ---------------- sqrt of min distance^2 ----------------
__global__ void finalize_kernel(const unsigned* __restrict__ psmin,
                                float* __restrict__ ps, int Q) {
    int i = blockIdx.x * blockDim.x + threadIdx.x;
    if (i < Q) ps[i] = sqrtf(__uint_as_float(psmin[i]));
}

// ---------------- per-image max score ----------------
__global__ void image_score_kernel(const float* __restrict__ ps,
                                   float* __restrict__ out, int P) {
    __shared__ float red[8];
    int b = blockIdx.x;
    int tid = threadIdx.x;
    float m = -3.4e38f;
    for (int i = tid; i < P; i += 256) m = fmaxf(m, ps[b * P + i]);
    for (int off = 16; off > 0; off >>= 1) m = fmaxf(m, __shfl_xor(m, off, 32));
    if ((tid & 31) == 0) red[tid >> 5] = m;
    __syncthreads();
    if (tid == 0) {
        float r = red[0];
        for (int i = 1; i < 8; ++i) r = fmaxf(r, red[i]);
        out[b] = r;
    }
}

// ---------------- bilinear 28x28 -> 224x224 (half-pixel, edge clamp) ----------------
__global__ void resize_kernel(const float* __restrict__ ps,
                              float* __restrict__ outm) {
    int idx = blockIdx.x * blockDim.x + threadIdx.x;
    if (idx >= NB * OH * OW) return;
    int x = idx % OW;
    int y = (idx / OW) % OH;
    int b = idx / (OH * OW);
    float sx = (x + 0.5f) * ((float)GW / OW) - 0.5f;
    float sy = (y + 0.5f) * ((float)GH / OH) - 0.5f;
    int x0 = (int)floorf(sx); float fx = sx - (float)x0;
    int y0 = (int)floorf(sy); float fy = sy - (float)y0;
    int x0c = x0 < 0 ? 0 : (x0 > GW - 1 ? GW - 1 : x0);
    int x1c = x0 + 1 < 0 ? 0 : (x0 + 1 > GW - 1 ? GW - 1 : x0 + 1);
    int y0c = y0 < 0 ? 0 : (y0 > GH - 1 ? GH - 1 : y0);
    int y1c = y0 + 1 < 0 ? 0 : (y0 + 1 > GH - 1 ? GH - 1 : y0 + 1);
    const float* g = ps + b * GH * GW;
    float v00 = g[y0c * GW + x0c], v01 = g[y0c * GW + x1c];
    float v10 = g[y1c * GW + x0c], v11 = g[y1c * GW + x1c];
    float v0 = v00 + (v01 - v00) * fx;
    float v1 = v10 + (v11 - v10) * fx;
    outm[idx] = v0 + (v1 - v0) * fy;
}

// np.pad mode='reflect' (no edge duplication): -i -> i, n-1+i -> n-1-i
__device__ inline int reflect_idx(int i, int n) {
    if (i < 0) i = -i;
    if (i >= n) i = 2 * n - 2 - i;
    return i;
}

__global__ void blur_w_kernel(const float* __restrict__ in,
                              float* __restrict__ out,
                              const float* __restrict__ w) {
    int idx = blockIdx.x * blockDim.x + threadIdx.x;
    if (idx >= NB * OH * OW) return;
    int x = idx % OW;
    const float* row = in + (idx - x);
    float s = 0.0f;
#pragma unroll
    for (int j = -RADIUS; j <= RADIUS; ++j)
        s += w[j + RADIUS] * row[reflect_idx(x + j, OW)];
    out[idx] = s;
}

__global__ void blur_h_kernel(const float* __restrict__ in,
                              float* __restrict__ out,
                              const float* __restrict__ w) {
    int idx = blockIdx.x * blockDim.x + threadIdx.x;
    if (idx >= NB * OH * OW) return;
    int x = idx % OW;
    int y = (idx / OW) % OH;
    int b = idx / (OH * OW);
    const float* col = in + (size_t)b * OH * OW + x;
    float s = 0.0f;
#pragma unroll
    for (int j = -RADIUS; j <= RADIUS; ++j)
        s += w[j + RADIUS] * col[reflect_idx(y + j, OH) * OW];
    out[idx] = s;
}

// ---------------------------------------------------------------------------
extern "C" void kernel_launch(void* const* d_in, const int* in_sizes, int n_in,
                              void* d_out, int out_size, void* d_ws, size_t ws_size,
                              hipStream_t stream) {
    const float* coreset  = (const float*)d_in[0];
    const float* features = (const float*)d_in[1];
    const int M = in_sizes[0] / D_DIM;   // 50000
    const int Q = in_sizes[1] / D_DIM;   // 3136
    float* out = (float*)d_out;          // [4 scores | 4*224*224 masks]
    const int NPIX = NB * OH * OW;

    // ---- workspace carve-up (256B aligned) ----
    size_t ofs = 0;
    char* base = (char*)d_ws;
    auto carve = [&](size_t bytes) -> void* {
        void* r = base + ofs;
        ofs = (ofs + bytes + 255) & ~(size_t)255;
        return r;
    };
    float*    a2      = (float*)carve((size_t)M * 4);
    float*    b2      = (float*)carve((size_t)Q * 4);
    unsigned* psmin   = (unsigned*)carve((size_t)Q * 4);
    float*    psf     = (float*)carve((size_t)Q * 4);
    float*    gw      = (float*)carve(64 * 4);
    float*    resized = (float*)carve((size_t)NPIX * 4);
    float*    tmpb    = (float*)carve((size_t)NPIX * 4);

    const size_t bbytes = (size_t)Q * D_DIM * 2;   // one bf16 features plane
    const size_t abytes = (size_t)M * D_DIM * 2;   // one bf16 coreset plane
    __bf16 *Bhi = nullptr, *Blo = nullptr, *Ahi = nullptr, *Alo = nullptr;
    bool haveB = false, haveA = false;
    if (ofs + 2 * bbytes + 4096 <= ws_size) {
        Bhi = (__bf16*)carve(bbytes);
        Blo = (__bf16*)carve(bbytes);
        haveB = true;
        if (ofs + 2 * abytes + 4096 <= ws_size) {
            Ahi = (__bf16*)carve(abytes);
            Alo = (__bf16*)carve(abytes);
            haveA = true;
        }
    }

    // ---- small prep ----
    init_min_kernel<<<(Q + 255) / 256, 256, 0, stream>>>(psmin, Q);
    gauss_kernel<<<1, 32, 0, stream>>>(gw);
    rownorm_kernel<<<(M + 7) / 8, 256, 0, stream>>>(coreset, a2, M);
    rownorm_kernel<<<(Q + 7) / 8, 256, 0, stream>>>(features, b2, Q);

    // ---- bf16 hi/lo pre-split (streaming, ~18us for coreset at 23.3 TB/s) ----
    if (haveB) {
        int n4 = (Q * D_DIM) / 4;
        split_bf16_kernel<<<(n4 + 255) / 256, 256, 0, stream>>>(features, Bhi, Blo, n4);
    }
    if (haveA) {
        int n4 = (M * D_DIM) / 4;
        split_bf16_kernel<<<(n4 + 255) / 256, 256, 0, stream>>>(coreset, Ahi, Alo, n4);
    }

    // ---- main GEMM + min ----
    dim3 grid(Q / 64, (M + 255) / 256);  // 49 x 196
    if (haveA) {
        gemm_min_kernel<2><<<grid, 256, 0, stream>>>(
            coreset, features, Ahi, Alo, Bhi, Blo, a2, b2, psmin, M);
    } else if (haveB) {
        gemm_min_kernel<1><<<grid, 256, 0, stream>>>(
            coreset, features, nullptr, nullptr, Bhi, Blo, a2, b2, psmin, M);
    } else {
        gemm_min_kernel<0><<<grid, 256, 0, stream>>>(
            coreset, features, nullptr, nullptr, nullptr, nullptr, a2, b2, psmin, M);
    }

    // ---- postprocessing ----
    finalize_kernel<<<(Q + 255) / 256, 256, 0, stream>>>(psmin, psf, Q);
    image_score_kernel<<<NB, 256, 0, stream>>>(psf, out, Q / NB);
    resize_kernel<<<(NPIX + 255) / 256, 256, 0, stream>>>(psf, resized);
    blur_w_kernel<<<(NPIX + 255) / 256, 256, 0, stream>>>(resized, tmpb, gw);
    blur_h_kernel<<<(NPIX + 255) / 256, 256, 0, stream>>>(tmpb, out + 4, gw);
}
